// i3Model_22686017257881
// MI455X (gfx1250) — compile-verified, matrix operand order checked by reading, not run on hardware
//
#include <hip/hip_runtime.h>
#include <hip/hip_bf16.h>

// ---------------------------------------------------------------------------
// Types for CDNA5 WMMA
// ---------------------------------------------------------------------------
typedef __attribute__((ext_vector_type(16))) __bf16 v16bf;
typedef __attribute__((ext_vector_type(8)))  float  v8f;
typedef __attribute__((ext_vector_type(4)))  int    v4i;

union FragB16 { v16bf v; uint4 q[2]; };

#define BM 128
#define BN 128
#define BK 32
#define LDK 40   // padded K-stride (elements) in LDS to dodge bank conflicts

__device__ __forceinline__ float gelu_exact(float x) {
    return 0.5f * x * (1.0f + erff(x * 0.70710678118654752f));
}

// ---------------------------------------------------------------------------
// CDNA5 async global->LDS copy (ASYNCcnt path) with toolchain fallbacks
// Builtin signature (from compiler diagnostic): (v4i AS1*, v4i AS3*, Ii, Ii)
// ---------------------------------------------------------------------------
__device__ __forceinline__ void async_copy_b128(const void* g, void* l) {
#if __has_builtin(__builtin_amdgcn_global_load_async_to_lds_b128)
    __builtin_amdgcn_global_load_async_to_lds_b128(
        (__attribute__((address_space(1))) v4i*)g,
        (__attribute__((address_space(3))) v4i*)l, 0, 0);
#else
    unsigned loff = (unsigned)(uintptr_t)(__attribute__((address_space(3))) void*)l;
    asm volatile("global_load_async_to_lds_b128 %0, %1, off"
                 :: "v"(loff), "v"(g) : "memory");
#endif
}

__device__ __forceinline__ void async_wait0() {
#if __has_builtin(__builtin_amdgcn_s_wait_asynccnt)
    __builtin_amdgcn_s_wait_asynccnt(0);
#else
    asm volatile("s_wait_asynccnt 0x0" ::: "memory");
#endif
}

// ---------------------------------------------------------------------------
// Weight pre-conversion: fp32 -> bf16, output [Npad][K] row-major,
// pad rows (n >= N) zero-filled so the GEMM needs no N-guards on B.
//   k_cvt_t: source is [K,N] (transpose while converting)
//   k_cvt_n: source is already [N,K]
// ---------------------------------------------------------------------------
__global__ __launch_bounds__(256) void k_cvt_t(
    const float* __restrict__ in, __bf16* __restrict__ out,
    int N, int K, int Npad)
{
    const int g = blockIdx.x * 256 + threadIdx.x;
    if (g >= Npad * K) return;
    const int n = g / K, k = g - n * K;
    const float v = (n < N) ? in[(size_t)k * N + n] : 0.f;
    out[g] = (__bf16)v;
}

__global__ __launch_bounds__(256) void k_cvt_n(
    const float* __restrict__ in, __bf16* __restrict__ out,
    int N, int K, int Npad)
{
    const int g = blockIdx.x * 256 + threadIdx.x;
    if (g >= Npad * K) return;
    const int n = g / K;
    const float v = (n < N) ? in[g] : 0.f;
    out[g] = (__bf16)v;
}

// ---------------------------------------------------------------------------
// WMMA GEMM: C[M,N] = act(A[M,K] @ W^T + bias) + resid
//   A:  fp32 row-major [M,K] (converted to bf16 while staging via VALU)
//   Bw: bf16 [Npad,K] row-major (pre-converted weights; async-DMA'd to LDS)
//   Double-buffered LDS; one barrier per K-step.
//   Requires M%128==0, K%32==0 (true for this model).
// ---------------------------------------------------------------------------
__global__ __launch_bounds__(256) void k_gemm(
    const float* __restrict__ A, const __bf16* __restrict__ Bw,
    const float* __restrict__ bias, const float* __restrict__ resid,
    float* __restrict__ C, int M, int N, int K, int actGelu)
{
    __shared__ __bf16 As[2][BM * LDK];
    __shared__ __bf16 Bs[2][BN * LDK];   // stored [n][k]

    const int tid   = threadIdx.x;
    const int lane  = tid & 31;
    const int wave  = tid >> 5;
    const int wm    = wave >> 2;      // 0..1 : 64-row strip
    const int wn    = wave & 3;       // 0..3 : 32-col strip
    const int lrow  = lane & 15;
    const int lhalf = lane >> 4;
    const int kbase = lhalf * 8;      // ISA 16-bit A/B lane K-chunk base

    const int m0 = blockIdx.y * BM;
    const int n0 = blockIdx.x * BN;

    v8f acc[4][2] = {};

    // staging maps
    const int a_row  = tid >> 1;           // 0..127
    const int a_koff = (tid & 1) * 16;
    const int b_row0 = tid >> 1;           // chunk pair: rows, 16B chunks
    const int b_ch0  = (tid & 1) * 2;      // chunks {0,1} or {2,3} of 4

    float4 areg[4];

    // ---- helpers -----------------------------------------------------------
    auto loadA = [&](int k0) {
        const float* src = A + (size_t)(m0 + a_row) * K + k0 + a_koff;
        #pragma unroll
        for (int j = 0; j < 4; ++j) areg[j] = *(const float4*)(src + j * 4);
        if (k0 + BK < K) __builtin_prefetch(src + BK, 0, 1);
    };
    auto storeA = [&](int buf) {
        __bf16* dst = &As[buf][a_row * LDK + a_koff];
        #pragma unroll
        for (int j = 0; j < 4; ++j) {
            dst[j * 4 + 0] = (__bf16)areg[j].x;
            dst[j * 4 + 1] = (__bf16)areg[j].y;
            dst[j * 4 + 2] = (__bf16)areg[j].z;
            dst[j * 4 + 3] = (__bf16)areg[j].w;
        }
    };
    auto stageB = [&](int k0, int buf) {   // 2 wave-level async b128 copies
        #pragma unroll
        for (int c = 0; c < 2; ++c) {
            const int ch = b_ch0 + c;      // 0..3 : which 16B chunk of the row
            const __bf16* g = Bw + (size_t)(n0 + b_row0) * K + k0 + ch * 8;
            async_copy_b128(g, &Bs[buf][b_row0 * LDK + ch * 8]);
        }
    };

    // ---- prologue: stage tile 0 --------------------------------------------
    stageB(0, 0);
    loadA(0);
    storeA(0);
    async_wait0();
    __syncthreads();

    const int nk = K / BK;
    for (int ki = 0; ki < nk; ++ki) {
        const int cur  = ki & 1;
        const int nxt  = cur ^ 1;
        const bool more = (ki + 1 < nk);

        // kick off next tile's DMA + global A loads before computing
        if (more) {
            stageB((ki + 1) * BK, nxt);
            loadA((ki + 1) * BK);
        }

        // ---- fragments: two b128 LDS reads per 16x32 tile per lane ----
        FragB16 af[4], bf[2];
        #pragma unroll
        for (int r = 0; r < 4; ++r) {
            const int row = wm * 64 + r * 16 + lrow;
            af[r].q[0] = *(const uint4*)(&As[cur][row * LDK + kbase]);
            af[r].q[1] = *(const uint4*)(&As[cur][row * LDK + 16 + kbase]);
        }
        #pragma unroll
        for (int c = 0; c < 2; ++c) {
            const int col = wn * 32 + c * 16 + lrow;
            bf[c].q[0] = *(const uint4*)(&Bs[cur][col * LDK + kbase]);
            bf[c].q[1] = *(const uint4*)(&Bs[cur][col * LDK + 16 + kbase]);
        }
        #pragma unroll
        for (int r = 0; r < 4; ++r)
            #pragma unroll
            for (int c = 0; c < 2; ++c)
                acc[r][c] = __builtin_amdgcn_wmma_f32_16x16x32_bf16(
                    false, af[r].v, false, bf[c].v, (short)0, acc[r][c],
                    false, false);

        if (more) {
            storeA(nxt);       // ds stores to the other buffer
            async_wait0();     // next tile's B DMA complete (this wave)
        }
        __syncthreads();       // all waves done reading cur / staging nxt
    }

    // ---- epilogue: bias / gelu / residual, fused ----
    #pragma unroll
    for (int r = 0; r < 4; ++r) {
        #pragma unroll
        for (int c = 0; c < 2; ++c) {
            const int col = n0 + wn * 32 + c * 16 + lrow;
            if (col < N) {
                #pragma unroll
                for (int v = 0; v < 8; ++v) {
                    const int row = m0 + wm * 64 + r * 16 + v + 8 * lhalf;
                    float f = acc[r][c][v];
                    if (bias)    f += bias[col];
                    if (actGelu) f = gelu_exact(f);
                    if (resid)   f += resid[(size_t)row * N + col];
                    C[(size_t)row * N + col] = f;
                }
            }
        }
    }
}

// ---------------------------------------------------------------------------
// Embedding gather: x[row,:] = embed[idx[row],:] + pos[row%512,:]
// ---------------------------------------------------------------------------
__global__ __launch_bounds__(256) void k_embed(
    const int* __restrict__ idx, const float* __restrict__ emb,
    const float* __restrict__ pos, float* __restrict__ X)
{
    const int row = blockIdx.x, tid = threadIdx.x;
    const int tok = idx[row];
    const int t   = row & 511;
    const float* e = emb + (size_t)tok * 768;
    const float* p = pos + (size_t)t * 768;
    float* x = X + (size_t)row * 768;
    x[tid]       = e[tid]       + p[tid];
    x[tid + 256] = e[tid + 256] + p[tid + 256];
    x[tid + 512] = e[tid + 512] + p[tid + 512];
}

// ---------------------------------------------------------------------------
// LayerNorm over D=768, one row per block
// ---------------------------------------------------------------------------
__global__ __launch_bounds__(256) void k_ln(
    const float* __restrict__ X, const float* __restrict__ s,
    const float* __restrict__ b, float* __restrict__ O)
{
    __shared__ float rs[256], rq[256];
    const int row = blockIdx.x, tid = threadIdx.x;
    const float* x = X + (size_t)row * 768;
    const float v0 = x[tid], v1 = x[tid + 256], v2 = x[tid + 512];
    rs[tid] = v0 + v1 + v2;
    rq[tid] = v0 * v0 + v1 * v1 + v2 * v2;
    __syncthreads();
    for (int off = 128; off > 0; off >>= 1) {
        if (tid < off) { rs[tid] += rs[tid + off]; rq[tid] += rq[tid + off]; }
        __syncthreads();
    }
    const float mean = rs[0] * (1.f / 768.f);
    const float var  = rq[0] * (1.f / 768.f) - mean * mean;
    const float rstd = rsqrtf(var + 1e-5f);
    float* o = O + (size_t)row * 768;
    o[tid]       = (v0 - mean) * rstd * s[tid]       + b[tid];
    o[tid + 256] = (v1 - mean) * rstd * s[tid + 256] + b[tid + 256];
    o[tid + 512] = (v2 - mean) * rstd * s[tid + 512] + b[tid + 512];
}

// ---------------------------------------------------------------------------
// State recurrence s_t = A s_{t-1} + u_t, DS=64, one wave32 per batch.
// Lane i holds s[i], s[i+32]; broadcast via shuffles (no barriers in chain).
// ---------------------------------------------------------------------------
__global__ __launch_bounds__(32) void k_scan(
    const float* __restrict__ u, const float* __restrict__ A,
    float* __restrict__ S)
{
    __shared__ float As[64 * 64];
    const int b = blockIdx.x, i = threadIdx.x;
    for (int t = i; t < 64 * 64; t += 32) As[t] = A[t];
    __syncthreads();
    float s0 = 0.f, s1 = 0.f;
    for (int t = 0; t < 512; ++t) {
        float acc0 = u[((size_t)(b * 512 + t)) * 64 + i];
        float acc1 = u[((size_t)(b * 512 + t)) * 64 + i + 32];
        #pragma unroll
        for (int j = 0; j < 32; ++j) {
            const float sj0 = __shfl(s0, j, 32);
            const float sj1 = __shfl(s1, j, 32);
            acc0 += As[i * 64 + j] * sj0 + As[i * 64 + 32 + j] * sj1;
            acc1 += As[(i + 32) * 64 + j] * sj0 + As[(i + 32) * 64 + 32 + j] * sj1;
        }
        s0 = acc0; s1 = acc1;
        S[((size_t)(b * 512 + t)) * 64 + i]      = s0;
        S[((size_t)(b * 512 + t)) * 64 + i + 32] = s1;
    }
}

// ---------------------------------------------------------------------------
// Token mix: h_t = w*h_{t-1} + (1-w)*x_t  (parallel over B*D, serial over T)
// ---------------------------------------------------------------------------
__global__ __launch_bounds__(256) void k_mix(
    const float* __restrict__ XN, const float* __restrict__ wmix,
    float* __restrict__ H)
{
    const int g = blockIdx.x * 256 + threadIdx.x;     // 0..3071
    const int b = g / 768, d = g % 768;
    const float w = wmix[d];
    float h = 0.f;
    const float* src = XN + (size_t)b * 512 * 768 + d;
    float* dst = H + (size_t)b * 512 * 768 + d;
    for (int t = 0; t < 512; ++t) {
        h = w * h + (1.f - w) * src[(size_t)t * 768];
        dst[(size_t)t * 768] = h;
    }
}

// X[m,n] += H[m,n] * Dv[n]
__global__ __launch_bounds__(256) void k_fma_rowvec(
    float* __restrict__ X, const float* __restrict__ H,
    const float* __restrict__ Dv, int total)
{
    const int g = blockIdx.x * 256 + threadIdx.x;
    if (g < total) X[g] += H[g] * Dv[g % 768];
}

// ---------------------------------------------------------------------------
// Causal attention, online softmax. One block per (b, h, 256-query tile).
// K/V tiles (64x64) staged in LDS; each thread owns one query.
// ---------------------------------------------------------------------------
__global__ __launch_bounds__(256) void k_attn(
    const float* __restrict__ QKV, float* __restrict__ O)
{
    __shared__ float Kt[64][65];
    __shared__ float Vt[64][65];
    const int qt  = blockIdx.x & 1;
    const int h   = (blockIdx.x >> 1) % 12;
    const int b   = blockIdx.x / 24;
    const int tid = threadIdx.x;
    const int q   = qt * 256 + tid;
    const size_t rstr = 2304;

    const float* qptr = QKV + ((size_t)(b * 512 + q)) * rstr + h * 64;
    float qv[64], o[64];
    #pragma unroll
    for (int d = 0; d < 64; ++d) { qv[d] = qptr[d]; o[d] = 0.f; }
    float m = -3.0e38f, l = 0.f;

    const int nkt = qt * 4 + 4;
    const int lr = tid >> 2, lc = (tid & 3) * 16;
    for (int kt = 0; kt < nkt; ++kt) {
        __syncthreads();
        const float* kp = QKV + ((size_t)(b * 512 + kt * 64 + lr)) * rstr + 768 + h * 64 + lc;
        const float* vp = kp + 768;
        #pragma unroll
        for (int j = 0; j < 4; ++j) {
            float4 kf = *(const float4*)(kp + j * 4);
            float4 vf = *(const float4*)(vp + j * 4);
            Kt[lr][lc + j * 4 + 0] = kf.x; Kt[lr][lc + j * 4 + 1] = kf.y;
            Kt[lr][lc + j * 4 + 2] = kf.z; Kt[lr][lc + j * 4 + 3] = kf.w;
            Vt[lr][lc + j * 4 + 0] = vf.x; Vt[lr][lc + j * 4 + 1] = vf.y;
            Vt[lr][lc + j * 4 + 2] = vf.z; Vt[lr][lc + j * 4 + 3] = vf.w;
        }
        __syncthreads();
        const int kmax = q - kt * 64;
        if (kmax < 0) continue;                 // after both barriers: safe
        const int klim = kmax < 63 ? kmax : 63;
        for (int kk = 0; kk <= klim; ++kk) {
            float sd = 0.f;
            #pragma unroll
            for (int d = 0; d < 64; ++d) sd += qv[d] * Kt[kk][d];
            sd *= 0.125f;                       // 1/sqrt(64)
            const float mn   = fmaxf(m, sd);
            const float corr = __expf(m - mn);
            const float p    = __expf(sd - mn);
            l = l * corr + p;
            #pragma unroll
            for (int d = 0; d < 64; ++d) o[d] = o[d] * corr + p * Vt[kk][d];
            m = mn;
        }
    }
    const float inv = 1.f / l;
    float* op = O + ((size_t)(b * 512 + q)) * 768 + h * 64;
    #pragma unroll
    for (int d = 0; d < 64; ++d) op[d] = o[d] * inv;
}

// ---------------------------------------------------------------------------
// Host-side orchestration
// ---------------------------------------------------------------------------
extern "C" void kernel_launch(void* const* d_in, const int* in_sizes, int n_in,
                              void* d_out, int out_size, void* d_ws, size_t ws_size,
                              hipStream_t stream)
{
    (void)in_sizes; (void)n_in; (void)out_size; (void)ws_size;

    const int*   idx     = (const int*)  d_in[0];
    const float* embed   = (const float*)d_in[1];
    const float* pos     = (const float*)d_in[2];
    const float* h_ln1_s = (const float*)d_in[3];
    const float* h_ln1_b = (const float*)d_in[4];
    const float* h_wmix  = (const float*)d_in[5];
    const float* h_A     = (const float*)d_in[6];
    const float* h_B     = (const float*)d_in[7];
    const float* h_C     = (const float*)d_in[8];
    const float* h_D     = (const float*)d_in[9];
    const float* h_ln2_s = (const float*)d_in[10];
    const float* h_ln2_b = (const float*)d_in[11];
    const float* h_w1    = (const float*)d_in[12];
    const float* h_b1    = (const float*)d_in[13];
    const float* h_w2    = (const float*)d_in[14];
    const float* h_b2    = (const float*)d_in[15];
    const float* a_ln1_s = (const float*)d_in[16];
    const float* a_ln1_b = (const float*)d_in[17];
    const float* a_qkv_w = (const float*)d_in[18];
    const float* a_qkv_b = (const float*)d_in[19];
    const float* a_out_w = (const float*)d_in[20];
    const float* a_out_b = (const float*)d_in[21];
    const float* a_ln2_s = (const float*)d_in[22];
    const float* a_ln2_b = (const float*)d_in[23];
    const float* a_w1    = (const float*)d_in[24];
    const float* a_b1    = (const float*)d_in[25];
    const float* a_w2    = (const float*)d_in[26];
    const float* a_b2    = (const float*)d_in[27];
    const float* lnf_s   = (const float*)d_in[28];
    const float* lnf_b   = (const float*)d_in[29];
    const float* head_w  = (const float*)d_in[30];
    const float* head_b  = (const float*)d_in[31];

    const int Mr = 2048;  // B*T
    float* ws = (float*)d_ws;
    float* x  = ws;                               // [2048,768]
    float* xn = x  + (size_t)Mr * 768;            // [2048,768]
    float* t0 = xn + (size_t)Mr * 768;            // [2048,3072] (also qkv)
    float* u  = t0 + (size_t)Mr * 3072;           // [2048,64]
    float* S  = u  + (size_t)Mr * 64;             // [2048,64]
    float* Hm = S  + (size_t)Mr * 64;             // [2048,768]
    float* aO = Hm + (size_t)Mr * 768;            // [2048,768]
    __bf16* wbuf = (__bf16*)(aO + (size_t)Mr * 768);  // [<=32000 x 768] bf16

    // GEMM wrapper: convert weight to bf16 [Npad,K] (pad rows zeroed), then
    // run the async-LDS / WMMA GEMM.  wTrans: weight source is [K,N].
    auto gemm = [&](const float* A, const float* W, int wTrans,
                    const float* bias, const float* res, float* Cp,
                    int M, int N, int K, int act) {
        const int Npad = ((N + 127) / 128) * 128;
        const int tot  = Npad * K;
        if (wTrans) k_cvt_t<<<(tot + 255) / 256, 256, 0, stream>>>(W, wbuf, N, K, Npad);
        else        k_cvt_n<<<(tot + 255) / 256, 256, 0, stream>>>(W, wbuf, N, K, Npad);
        dim3 g(Npad / 128, M / 128);
        k_gemm<<<g, 256, 0, stream>>>(A, wbuf, bias, res, Cp, M, N, K, act);
    };

    k_embed<<<Mr, 256, 0, stream>>>(idx, embed, pos, x);

    // ---- hybrid (RWKV/Mamba-style) blocks ----
    for (int i = 0; i < 10; ++i) {
        k_ln<<<Mr, 256, 0, stream>>>(x, h_ln1_s + i * 768, h_ln1_b + i * 768, xn);
        gemm(xn, h_B + (size_t)i * 64 * 768, 0, nullptr, nullptr, u, Mr, 64, 768, 0);
        k_scan<<<4, 32, 0, stream>>>(u, h_A + (size_t)i * 64 * 64, S);
        k_mix<<<12, 256, 0, stream>>>(xn, h_wmix + i * 768, Hm);
        gemm(S, h_C + (size_t)i * 768 * 64, 0, nullptr, x, x, Mr, 768, 64, 0);
        k_fma_rowvec<<<6144, 256, 0, stream>>>(x, Hm, h_D + i * 768, Mr * 768);
        k_ln<<<Mr, 256, 0, stream>>>(x, h_ln2_s + i * 768, h_ln2_b + i * 768, xn);
        gemm(xn, h_w1 + (size_t)i * 768 * 3072, 1, h_b1 + i * 3072, nullptr, t0, Mr, 3072, 768, 1);
        gemm(t0, h_w2 + (size_t)i * 3072 * 768, 1, h_b2 + i * 768, x, x, Mr, 768, 3072, 0);
    }
    // ---- full attention blocks ----
    for (int i = 0; i < 6; ++i) {
        k_ln<<<Mr, 256, 0, stream>>>(x, a_ln1_s + i * 768, a_ln1_b + i * 768, xn);
        gemm(xn, a_qkv_w + (size_t)i * 768 * 2304, 1, a_qkv_b + i * 2304, nullptr, t0, Mr, 2304, 768, 0);
        k_attn<<<96, 256, 0, stream>>>(t0, aO);
        gemm(aO, a_out_w + (size_t)i * 768 * 768, 1, a_out_b + i * 768, x, x, Mr, 768, 768, 0);
        k_ln<<<Mr, 256, 0, stream>>>(x, a_ln2_s + i * 768, a_ln2_b + i * 768, xn);
        gemm(xn, a_w1 + (size_t)i * 768 * 3072, 1, a_b1 + i * 3072, nullptr, t0, Mr, 3072, 768, 1);
        gemm(t0, a_w2 + (size_t)i * 3072 * 768, 1, a_b2 + i * 768, x, x, Mr, 768, 3072, 0);
    }
    // ---- final LN + vocab head ----
    k_ln<<<Mr, 256, 0, stream>>>(x, lnf_s, lnf_b, xn);
    gemm(xn, head_w, 1, head_b, nullptr, (float*)d_out, Mr, 32000, 768, 0);
}
